// CrossLevelAttention_42545946034839
// MI455X (gfx1250) — matrix operand order repaired and checked by reading, not compiled, not run on hardware
//
#include <hip/hip_runtime.h>
#include <hip/hip_bf16.h>

typedef __attribute__((ext_vector_type(16))) _Float16 v16h;
typedef __attribute__((ext_vector_type(8)))  _Float16 v8h;
typedef __attribute__((ext_vector_type(2)))  _Float16 v2h;
typedef __attribute__((ext_vector_type(2)))  __fp16   v2fp;
typedef __attribute__((ext_vector_type(8)))  float    v8f;

#define N_HEAD   16
#define HEAD_DIM 64
#define TQ       2048
#define CDIM     (N_HEAD * HEAD_DIM)
#define SCALE    0.125f
#define LOG2E    1.44269504088896340736f
#define NEG_BIG  (-3.0e38f)

#if __has_builtin(__builtin_amdgcn_exp2f)
#define EXP2(x) __builtin_amdgcn_exp2f(x)
#else
#define EXP2(x) exp2f(x)
#endif

#if __has_builtin(__builtin_amdgcn_cvt_pkrtz)
__device__ __forceinline__ v2h pk_f16(float a, float b) {
    union { v2fp p; v2h h; } u;
    u.p = __builtin_amdgcn_cvt_pkrtz(a, b);
    return u.h;
}
#else
__device__ __forceinline__ v2h pk_f16(float a, float b) {
    v2h r; r[0] = (_Float16)a; r[1] = (_Float16)b; return r;
}
#endif

// butterfly max across 16-lane halves via DPP (no LDS traffic)
template <int CTRL>
__device__ __forceinline__ float dpp_fmax(float x) {
    union { float f; int i; } u, v;
    u.f = x;
    v.i = __builtin_amdgcn_update_dpp(0, u.i, CTRL, 0xf, 0xf, true);
    return fmaxf(x, v.f);
}
__device__ __forceinline__ float halfwave_max(float x) {
    x = dpp_fmax<0xB1>(x);    // quad_perm(1,0,3,2) : xor 1
    x = dpp_fmax<0x4E>(x);    // quad_perm(2,3,0,1) : xor 2
    x = dpp_fmax<0x141>(x);   // row_half_mirror    : xor 4
    x = dpp_fmax<0x140>(x);   // row_mirror         : xor 8
    return x;
}

// Block: 128 threads = 4 waves (wave32). Each wave owns 16 query rows,
// block cooperatively stages 32-key K/V tiles (double buffered).
__global__ __launch_bounds__(128) void cla_kernel(
    const float* __restrict__ Q,
    const float* __restrict__ K0, const float* __restrict__ V0,
    const float* __restrict__ K1, const float* __restrict__ V1,
    const float* __restrict__ K2, const float* __restrict__ V2,
    const float* __restrict__ logits,
    float* __restrict__ out)
{
    __shared__ __align__(32) _Float16 Kt[2][32 * 64];   // [buf][key*64 + dim]  row-major
    __shared__ __align__(32) _Float16 Vt[2][64 * 32];   // [buf][dim*32 + key]  transposed
    __shared__ __align__(32) _Float16 Pt[4][16 * 32];   // per-wave P tile, A-frag-contiguous layout

    const int qtile = blockIdx.x;     // 64-query tile
    const int h     = blockIdx.y;
    const int b     = blockIdx.z;
    const int tid   = threadIdx.x;
    const int wave  = tid >> 5;
    const int lane  = tid & 31;
    const int lx    = lane & 15;
    const int lh    = lane >> 4;

    // ---- level-weight softmax (uniform) ----
    const float g0 = logits[0], g1 = logits[1], g2 = logits[2];
    const float gm = fmaxf(g0, fmaxf(g1, g2));
    const float e0 = __expf(g0 - gm), e1 = __expf(g1 - gm), e2 = __expf(g2 - gm);
    const float einv = 1.0f / (e0 + e1 + e2);
    const float wlev[3] = { e0 * einv, e1 * einv, e2 * einv };

    // ---- Q A-fragments, pre-scaled by SCALE*log2(e) so S lands in exp2 domain ----
    const int q0w = qtile * 64 + wave * 16;
    const float qs = SCALE * LOG2E;
    v16h qa[2];
    {
        const float* qp = Q + (((size_t)b * N_HEAD + h) * TQ + (q0w + lx)) * HEAD_DIM;
        #pragma unroll
        for (int f = 0; f < 2; ++f) {
            v16h v;
            #pragma unroll
            for (int i = 0; i < 8; ++i) v[i]     = (_Float16)(qp[f * 32 + lh * 16 + i] * qs);
            #pragma unroll
            for (int i = 0; i < 8; ++i) v[8 + i] = (_Float16)(qp[f * 32 + lh * 16 + 8 + i] * qs);
            qa[f] = v;
        }
    }

    // all-ones B fragment (32x16) for the row-sum WMMA
    v16h onesB;
    #pragma unroll
    for (int i = 0; i < 16; ++i) onesB[i] = (_Float16)1.0f;

    v8f acc[4];
    #pragma unroll
    for (int nb = 0; nb < 4; ++nb)
        #pragma unroll
        for (int r = 0; r < 8; ++r) acc[nb][r] = 0.0f;

    const float* Ks[3]  = { K0, K1, K2 };
    const float* Vs[3]  = { V0, V1, V2 };
    const int    Tms[3] = { TQ, 512, 128 };

    // staging decomposition: thread -> key pair (2 keys) x 8 dims
    const int sp  = tid >> 3;          // key pair index 0..15 -> keys 2sp, 2sp+1
    const int sdb = (tid & 7) * 8;     // dim base

    // P-store position: natural key order so each lane's A-fragment is one
    // contiguous 32B read at Pt[m*32 + lh*16]
    const int pos0 = lx;               // key lx      -> pos
                                       // key 16+lx   -> pos0 + 16

    #pragma unroll   // full unroll: concrete K/V pointers (global AS) + causal specialization
    for (int lev = 0; lev < 3; ++lev) {
        const float* Kl = Ks[lev];
        const float* Vl = Vs[lev];
        const int Tm = Tms[lev];
        const int nT = (lev == 0) ? (qtile * 2 + 2) : (Tm / 32);  // causal: keys <= last row

        v8f Ol[4];
        float rmax[8], rsum[8];
        #pragma unroll
        for (int r = 0; r < 8; ++r) { rmax[r] = NEG_BIG; rsum[r] = 0.0f; }
        #pragma unroll
        for (int nb = 0; nb < 4; ++nb)
            #pragma unroll
            for (int r = 0; r < 8; ++r) Ol[nb][r] = 0.0f;

        int buf = 0;

        // ---- stage helper body (tile kt -> buffer ob) ----
        auto stage = [&](int kt, int ob) {
            const int gk = kt * 32 + sp * 2;
            const float* kp0 = Kl + ((size_t)b * Tm + gk) * CDIM + (size_t)h * HEAD_DIM + sdb;
            const float* kp1 = kp0 + CDIM;
            const float* vp0 = Vl + ((size_t)b * Tm + gk) * CDIM + (size_t)h * HEAD_DIM + sdb;
            const float* vp1 = vp0 + CDIM;
            union { v2h h2[4]; v8h h8; } k0u, k1u;
            #pragma unroll
            for (int i = 0; i < 4; ++i) {
                k0u.h2[i] = pk_f16(kp0[2 * i], kp0[2 * i + 1]);
                k1u.h2[i] = pk_f16(kp1[2 * i], kp1[2 * i + 1]);
            }
            *(v8h*)&Kt[ob][(sp * 2) * 64 + sdb]     = k0u.h8;
            *(v8h*)&Kt[ob][(sp * 2 + 1) * 64 + sdb] = k1u.h8;
            #pragma unroll
            for (int i = 0; i < 8; ++i) {
                *(v2h*)&Vt[ob][(sdb + i) * 32 + sp * 2] = pk_f16(vp0[i], vp1[i]);
            }
        };

        stage(0, 0);
        __syncthreads();

        for (int t = 0; t < nT; ++t) {
            if (t + 1 < nT) stage(t + 1, buf ^ 1);

            // ---- S = (Q*qs) * K^T : two 16x16 f32 tiles (keys 0-15, 16-31) ----
            v8f s0 = {0,0,0,0,0,0,0,0}, s1 = {0,0,0,0,0,0,0,0};
            #pragma unroll
            for (int f = 0; f < 2; ++f) {
                v16h kb0 = *(const v16h*)&Kt[buf][( 0 + lx) * 64 + f * 32 + lh * 16];
                v16h kb1 = *(const v16h*)&Kt[buf][(16 + lx) * 64 + f * 32 + lh * 16];
                s0 = __builtin_amdgcn_wmma_f32_16x16x32_f16(false, qa[f], false, kb0,
                                                            (short)0, s0, false, false);
                s1 = __builtin_amdgcn_wmma_f32_16x16x32_f16(false, qa[f], false, kb1,
                                                            (short)0, s1, false, false);
            }

            // ---- causal mask: only tiles touching/over the diagonal (wave-uniform) ----
            const int k0 = t * 32;
            if (lev == 0 && k0 + 31 > q0w) {
                #pragma unroll
                for (int r = 0; r < 8; ++r) {
                    const int qg = q0w + r + lh * 8;
                    if (k0 + lx      > qg) s0[r] = NEG_BIG;
                    if (k0 + 16 + lx > qg) s1[r] = NEG_BIG;
                }
            }

            // ---- online softmax (exp2 domain); max via DPP, sum via WMMA below ----
            float corr8[8];
            #pragma unroll
            for (int r = 0; r < 8; ++r) {
                float mx = halfwave_max(fmaxf(s0[r], s1[r]));
                const float nm = fmaxf(rmax[r], mx);
                const float corr = EXP2(rmax[r] - nm);
                corr8[r] = corr;
                rmax[r] = nm;
                const float p0 = EXP2(s0[r] - nm);
                const float p1 = EXP2(s1[r] - nm);
                #pragma unroll
                for (int nb = 0; nb < 4; ++nb) Ol[nb][r] *= corr;
                const int m = r + lh * 8;
                Pt[wave][m * 32 + pos0]     = (_Float16)p0;
                Pt[wave][m * 32 + pos0 + 16] = (_Float16)p1;
            }

            // wave-private LDS RAW ordering (same-wave DS ops are in-order)
            asm volatile("s_wait_dscnt 0" ::: "memory");

            // ---- P A-fragment: one contiguous 32B read per lane ----
            v16h pa = *(const v16h*)&Pt[wave][lx * 32 + lh * 16];

            // row-sum via WMMA: D[m][*] = sum_k P[m][k], broadcast across lanes
            v8f ls = {0,0,0,0,0,0,0,0};
            ls = __builtin_amdgcn_wmma_f32_16x16x32_f16(false, pa, false, onesB,
                                                        (short)0, ls, false, false);

            // ---- O += P * V : 4 tiles across dh ----
            #pragma unroll
            for (int nb = 0; nb < 4; ++nb) {
                v16h vb = *(const v16h*)&Vt[buf][(nb * 16 + lx) * 32 + lh * 16];
                Ol[nb] = __builtin_amdgcn_wmma_f32_16x16x32_f16(false, pa, false, vb,
                                                                (short)0, Ol[nb], false, false);
            }

            #pragma unroll
            for (int r = 0; r < 8; ++r) rsum[r] = rsum[r] * corr8[r] + ls[r];

            __syncthreads();
            buf ^= 1;
        }

        // ---- merge level into weighted accumulator ----
        #pragma unroll
        for (int r = 0; r < 8; ++r) {
            const float s = wlev[lev] / rsum[r];
            #pragma unroll
            for (int nb = 0; nb < 4; ++nb) acc[nb][r] += Ol[nb][r] * s;
        }
        __syncthreads();
    }

    // ---- write out[b, q, h*64 + d]  (C/D layout: lane = d%16, vgpr = row) ----
    #pragma unroll
    for (int r = 0; r < 8; ++r) {
        const int q = q0w + r + lh * 8;
        #pragma unroll
        for (int nb = 0; nb < 4; ++nb) {
            out[((size_t)b * TQ + q) * CDIM + (size_t)h * HEAD_DIM + nb * 16 + lx] = acc[nb][r];
        }
    }
}

extern "C" void kernel_launch(void* const* d_in, const int* in_sizes, int n_in,
                              void* d_out, int out_size, void* d_ws, size_t ws_size,
                              hipStream_t stream) {
    const float* Q  = (const float*)d_in[0];
    const float* K0 = (const float*)d_in[1];
    const float* V0 = (const float*)d_in[2];
    const float* K1 = (const float*)d_in[3];
    const float* V1 = (const float*)d_in[4];
    const float* K2 = (const float*)d_in[5];
    const float* V2 = (const float*)d_in[6];
    const float* lg = (const float*)d_in[7];
    float* out = (float*)d_out;

    const int B = in_sizes[0] / (N_HEAD * TQ * HEAD_DIM);
    dim3 grid(TQ / 64, N_HEAD, B);
    cla_kernel<<<grid, 128, 0, stream>>>(Q, K0, V0, K1, V1, K2, V2, lg, out);
}
